// Graph_Transformer_24197845746077
// MI455X (gfx1250) — compile-verified, matrix operand order checked by reading, not run on hardware
//
#include <hip/hip_runtime.h>
#include <hip/hip_bf16.h>

// ---------------------------------------------------------------------------
// Graph Transformer for MI455X (gfx1250).
// Dense matmuls -> v_wmma_f32_16x16x32_f16 (f16 operands, fp32 accumulate).
// Graph gather/scatter/softmax -> wave-per-edge VALU kernels + f32 atomics.
// ---------------------------------------------------------------------------

#define NN   50000
#define NE   800000
#define NG   256
#define D_HC 128
#define D_EF 64

typedef __attribute__((ext_vector_type(16))) _Float16 v16h;
typedef __attribute__((ext_vector_type(8)))  _Float16 v8h;
typedef __attribute__((ext_vector_type(8)))  float    v8f;
typedef __attribute__((ext_vector_type(4)))  float    v4f;

__device__ __forceinline__ float eluf(float x) { return x > 0.f ? x : __expf(x) - 1.f; }

// order-preserving float <-> uint for atomicMax on floats
__device__ __forceinline__ unsigned fenc(float x) {
  unsigned u = __float_as_uint(x);
  return (u & 0x80000000u) ? ~u : (u | 0x80000000u);
}
__device__ __forceinline__ float fdec(unsigned u) {
  u = (u & 0x80000000u) ? (u & 0x7fffffffu) : ~u;
  return __uint_as_float(u);
}

// ---------------------------------------------------------------------------
// WMMA GEMM: out[M x Nc] = A[M x K] @ W[K x Nc] (+bias) (+ELU)
// 256 threads = 8 waves; wave w -> rows [blk*128 + w*16, +16); Nc/16 col tiles.
// W is staged transposed into LDS as f16 so B fragments are two contiguous
// 16B DS loads per lane (ISA 16-bit B-matrix layout).
// ---------------------------------------------------------------------------
template <int K, int Nc, int ACT>
__global__ void __launch_bounds__(256) gemm_wmma(const float* __restrict__ A,
                                                 const float* __restrict__ W,
                                                 const float* __restrict__ bias,
                                                 float* __restrict__ out, int M) {
  constexpr int Kp = (K < 32) ? 32 : K;  // zero-pad K=16 up to one 32-chunk
  constexpr int NT = Nc / 16;
  __shared__ _Float16 Wt[Nc * Kp];  // transposed: Wt[n*Kp + k]

  for (int idx = threadIdx.x; idx < Kp * Nc; idx += 256) {
    int kk = idx / Nc, n = idx % Nc;  // coalesced global read of W
    float wv = (kk < K) ? W[(size_t)kk * Nc + n] : 0.f;
    Wt[n * Kp + kk] = (_Float16)wv;
  }
  __syncthreads();

  int lane = threadIdx.x & 31;
  int wave = threadIdx.x >> 5;
  int row0 = blockIdx.x * 128 + wave * 16;
  int arow = row0 + (lane & 15);
  if (arow >= M) arow = M - 1;  // clamp: keep EXEC all-ones through WMMA
  int klane = (lane >> 4) << 3;  // 0 or 8 (K-half per lane group)
  const float* Ar = A + (size_t)arow * K;

  v8f acc[NT];
  for (int t = 0; t < NT; ++t)
    for (int r = 0; r < 8; ++r) acc[t][r] = 0.f;

  for (int kc = 0; kc < K; kc += 32) {
    int kb = kc + klane;
    v4f x0 = {0.f, 0.f, 0.f, 0.f}, x1 = x0, x2 = x0, x3 = x0;
    if (kb < K)      x0 = *(const v4f*)(Ar + kb);
    if (kb + 4 < K)  x1 = *(const v4f*)(Ar + kb + 4);
    if (kb + 16 < K) x2 = *(const v4f*)(Ar + kb + 16);
    if (kb + 20 < K) x3 = *(const v4f*)(Ar + kb + 20);
    v16h a;
    a[0] = (_Float16)x0[0];  a[1] = (_Float16)x0[1];  a[2] = (_Float16)x0[2];  a[3] = (_Float16)x0[3];
    a[4] = (_Float16)x1[0];  a[5] = (_Float16)x1[1];  a[6] = (_Float16)x1[2];  a[7] = (_Float16)x1[3];
    a[8] = (_Float16)x2[0];  a[9] = (_Float16)x2[1];  a[10] = (_Float16)x2[2]; a[11] = (_Float16)x2[3];
    a[12] = (_Float16)x3[0]; a[13] = (_Float16)x3[1]; a[14] = (_Float16)x3[2]; a[15] = (_Float16)x3[3];
    for (int t = 0; t < NT; ++t) {
      int col = t * 16 + (lane & 15);
      const _Float16* wp = &Wt[col * Kp + kb];
      v8h blo = *(const v8h*)wp;          // K = kb .. kb+7
      v8h bhi = *(const v8h*)(wp + 16);   // K = kb+16 .. kb+23
      v16h b;
      for (int u = 0; u < 8; ++u) { b[u] = blo[u]; b[u + 8] = bhi[u]; }
      acc[t] = __builtin_amdgcn_wmma_f32_16x16x32_f16(false, a, false, b,
                                                      (short)0, acc[t], false, false);
    }
  }

  int mhalf = (lane >> 4) << 3;
  for (int t = 0; t < NT; ++t) {
    int col = t * 16 + (lane & 15);
    float bv = bias ? bias[col] : 0.f;
    for (int r = 0; r < 8; ++r) {
      int gr = row0 + mhalf + r;
      if (gr < M) {
        float y = acc[t][r] + bv;
        if (ACT) y = eluf(y);
        out[(size_t)gr * Nc + col] = y;
      }
    }
  }
}

// ---------------------------------------------------------------------------
// Utility fills
// ---------------------------------------------------------------------------
__global__ void fill_f32(float* p, float v, int n) {
  int i = blockIdx.x * 256 + threadIdx.x;
  if (i < n) p[i] = v;
}
__global__ void fill_u32(unsigned* p, unsigned v, int n) {
  int i = blockIdx.x * 256 + threadIdx.x;
  if (i < n) p[i] = v;
}
__global__ void elu_inplace(float* p, int n) {
  int i = blockIdx.x * 256 + threadIdx.x;
  if (i < n) p[i] = eluf(p[i]);
}

// ---------------------------------------------------------------------------
// graph LayerNorm (PyG mode='graph'): stats over all nodes+channels per graph
// ---------------------------------------------------------------------------
__global__ void gln_stats(const float* __restrict__ X, const int* __restrict__ batch,
                          float* gcnt, float* gsum, float* gss, int n) {
  int w = blockIdx.x * 8 + (threadIdx.x >> 5);
  int lane = threadIdx.x & 31;
  if (w >= n) return;
  const float* xp = X + (size_t)w * D_HC;
  float s = 0.f, ss = 0.f;
  for (int j = 0; j < 4; ++j) {
    float x = xp[lane + 32 * j];
    s += x; ss += x * x;
  }
  for (int m = 16; m; m >>= 1) { s += __shfl_xor(s, m, 32); ss += __shfl_xor(ss, m, 32); }
  if (lane == 0) {
    int b = batch[w];
    atomicAdd(gcnt + b, (float)D_HC);
    atomicAdd(gsum + b, s);
    atomicAdd(gss + b, ss);
  }
}

__global__ void gln_apply(const float* __restrict__ X, float* __restrict__ Y,
                          const int* __restrict__ batch, const float* gcnt,
                          const float* gsum, const float* gss,
                          const float* __restrict__ gamma, const float* __restrict__ beta,
                          int doElu, int total) {
  int i = blockIdx.x * 256 + threadIdx.x;
  if (i >= total) return;
  int nIdx = i >> 7, c = i & 127;
  int b = batch[nIdx];
  float cnt = fmaxf(gcnt[b], 1.f);
  float mean = gsum[b] / cnt;
  float var = fmaxf(gss[b] / cnt - mean * mean, 0.f);
  float y = (X[i] - mean) * rsqrtf(var + 1e-5f) * gamma[c] + beta[c];
  if (doElu) y = eluf(y);
  Y[i] = y;
}

// ---------------------------------------------------------------------------
// Edge attention: logits, softmax over dst, weighted aggregation
// ---------------------------------------------------------------------------
__global__ void edge_logits(const float* __restrict__ Q, const float* __restrict__ Kf,
                            const float* __restrict__ EP, const int* __restrict__ src,
                            const int* __restrict__ dst, float* __restrict__ alpha,
                            unsigned* __restrict__ nmax, int nE) {
  int e = blockIdx.x * 8 + (threadIdx.x >> 5);
  int lane = threadIdx.x & 31;
  if (e >= nE) return;
  int s = src[e], d = dst[e];
  const float* qp = Q + (size_t)d * D_HC;
  const float* kp = Kf + (size_t)s * D_HC;
  const float* ep = EP + (size_t)e * D_HC;
  float p0 = 0.f, p1 = 0.f;
  for (int j = 0; j < 2; ++j) {
    int c = lane + 32 * j;
    p0 += qp[c] * (kp[c] + ep[c]);              // head 0: dims 0..63
    p1 += qp[c + 64] * (kp[c + 64] + ep[c + 64]);  // head 1: dims 64..127
  }
  for (int m = 16; m; m >>= 1) { p0 += __shfl_xor(p0, m, 32); p1 += __shfl_xor(p1, m, 32); }
  if (lane == 0) {
    float a0 = p0 * 0.125f, a1 = p1 * 0.125f;  // / sqrt(C=64)
    alpha[e * 2 + 0] = a0;
    alpha[e * 2 + 1] = a1;
    atomicMax(nmax + d * 2 + 0, fenc(a0));
    atomicMax(nmax + d * 2 + 1, fenc(a1));
  }
}

__global__ void edge_exp(float* alpha, const int* __restrict__ dst,
                         const unsigned* __restrict__ nmax, float* nsum, int total) {
  int i = blockIdx.x * 256 + threadIdx.x;
  if (i >= total) return;
  int e = i >> 1, hh = i & 1;
  int d = dst[e];
  float ex = __expf(alpha[i] - fdec(nmax[d * 2 + hh]));
  alpha[i] = ex;
  atomicAdd(nsum + d * 2 + hh, ex);
}

__global__ void edge_aggr(const float* __restrict__ V, const float* __restrict__ EP,
                          const float* __restrict__ alpha, const float* __restrict__ nsum,
                          const int* __restrict__ src, const int* __restrict__ dst,
                          float* __restrict__ acc, int nE) {
  int e = blockIdx.x * 8 + (threadIdx.x >> 5);
  int lane = threadIdx.x & 31;
  if (e >= nE) return;
  int s = src[e], d = dst[e];
  float w0 = alpha[e * 2 + 0] / nsum[d * 2 + 0];
  float w1 = alpha[e * 2 + 1] / nsum[d * 2 + 1];
  const float* vp = V + (size_t)s * D_HC;
  const float* ep = EP + (size_t)e * D_HC;
  float* ap = acc + (size_t)d * D_HC;
  for (int j = 0; j < 4; ++j) {
    int c = lane + 32 * j;
    float wgt = (c < 64) ? w0 : w1;
    atomicAdd(ap + c, (vp[c] + ep[c]) * wgt);
  }
}

// ---------------------------------------------------------------------------
// SAG pooling pieces
// ---------------------------------------------------------------------------
__global__ void node_aggr(const float* __restrict__ Hn, const int* __restrict__ src,
                          const int* __restrict__ dst, float* __restrict__ aggr, int nE) {
  int e = blockIdx.x * 8 + (threadIdx.x >> 5);
  int lane = threadIdx.x & 31;
  if (e >= nE) return;
  int s = src[e], d = dst[e];
  for (int j = 0; j < 4; ++j) {
    int c = lane + 32 * j;
    atomicAdd(aggr + (size_t)d * D_HC + c, Hn[(size_t)s * D_HC + c]);
  }
}

__global__ void node_score(const float* __restrict__ aggr, const float* __restrict__ Hn,
                           const float* __restrict__ wrel, const float* __restrict__ brel,
                           const float* __restrict__ wroot, const int* __restrict__ batch,
                           float* __restrict__ sc, unsigned* __restrict__ gmax, int n) {
  int w = blockIdx.x * 8 + (threadIdx.x >> 5);
  int lane = threadIdx.x & 31;
  if (w >= n) return;
  float p = 0.f;
  for (int j = 0; j < 4; ++j) {
    int c = lane + 32 * j;
    p += aggr[(size_t)w * D_HC + c] * wrel[c] + Hn[(size_t)w * D_HC + c] * wroot[c];
  }
  for (int m = 16; m; m >>= 1) p += __shfl_xor(p, m, 32);
  if (lane == 0) {
    float v = p + brel[0];
    sc[w] = v;
    atomicMax(gmax + batch[w], fenc(v));
  }
}

__global__ void score_exp(float* sc, const int* __restrict__ batch,
                          const unsigned* __restrict__ gmax, float* gssum, int n) {
  int i = blockIdx.x * 256 + threadIdx.x;
  if (i >= n) return;
  int b = batch[i];
  float ex = __expf(sc[i] - fdec(gmax[b]));
  sc[i] = ex;
  atomicAdd(gssum + b, ex);
}

__global__ void emb_accum(const float* __restrict__ Hn, const float* __restrict__ sc,
                          const float* __restrict__ gssum, const int* __restrict__ batch,
                          float* __restrict__ emb, int total) {
  int i = blockIdx.x * 256 + threadIdx.x;
  if (i >= total) return;
  int nIdx = i >> 7, c = i & 127;
  int b = batch[nIdx];
  float wgt = sc[nIdx] / gssum[b];
  atomicAdd(emb + b * D_HC + c, Hn[i] * wgt);
}

__global__ void edge_seg_sum(const float* __restrict__ ea2, const int* __restrict__ src,
                             const int* __restrict__ batch, float* __restrict__ eagg,
                             int total) {
  int i = blockIdx.x * 256 + threadIdx.x;
  if (i >= total) return;
  int e = i >> 6, c = i & 63;
  int b = batch[src[e]];
  atomicAdd(eagg + b * D_EF + c, ea2[i]);
}

__global__ void emb_finalize(const float* __restrict__ emb, const float* __restrict__ embe,
                             float* __restrict__ out, int isFirst) {
  __shared__ float red[128];
  int g = blockIdx.x, c = threadIdx.x;
  float v = emb[g * D_HC + c] * embe[g * D_HC + c];
  red[c] = v * v;
  __syncthreads();
  for (int s2 = 64; s2; s2 >>= 1) {
    if (c < s2) red[c] += red[c + s2];
    __syncthreads();
  }
  float nrm = fmaxf(sqrtf(red[0]), 1e-12f);
  float y = eluf(v / nrm);
  if (isFirst) out[g * D_HC + c] = 0.6f * y;
  else out[g * D_HC + c] += 0.4f * y;
}

// ---------------------------------------------------------------------------
// Host orchestration
// ---------------------------------------------------------------------------
static inline int cdiv(int a, int b) { return (a + b - 1) / b; }

extern "C" void kernel_launch(void* const* d_in, const int* in_sizes, int n_in,
                              void* d_out, int out_size, void* d_ws, size_t ws_size,
                              hipStream_t stream) {
  const float* x         = (const float*)d_in[0];
  const float* edge_attr = (const float*)d_in[1];
  const int*   eidx      = (const int*)d_in[2];
  const int*   batch     = (const int*)d_in[3];
  const float* w0        = (const float*)d_in[4];
  const float* b0        = (const float*)d_in[5];
  const float* we0       = (const float*)d_in[6];
  const float* be0       = (const float*)d_in[7];
  const float* ln0_g     = (const float*)d_in[8];
  const float* ln0_b     = (const float*)d_in[9];
  const float* qkv_w     = (const float*)d_in[10];
  const float* qkv_b     = (const float*)d_in[11];
  const float* edge_w    = (const float*)d_in[12];
  const float* skip_w    = (const float*)d_in[13];
  const float* skip_b    = (const float*)d_in[14];
  const float* up_w      = (const float*)d_in[15];
  const float* up_b      = (const float*)d_in[16];
  const float* ng        = (const float*)d_in[17];
  const float* nbb       = (const float*)d_in[18];
  const float* sag_wrel  = (const float*)d_in[19];
  const float* sag_brel  = (const float*)d_in[20];
  const float* sag_wroot = (const float*)d_in[21];
  const float* re_w      = (const float*)d_in[22];
  const float* re_b      = (const float*)d_in[23];
  float* out = (float*)d_out;

  const int* src = eidx;
  const int* dst = eidx + NE;

  // workspace layout
  float* ws = (float*)d_ws;
  size_t off = 0;
  auto alloc = [&](size_t n) { float* p = ws + off; off += n; return p; };
  float*    h     = alloc((size_t)NN * D_HC);
  float*    acc   = alloc((size_t)NN * D_HC);
  float*    qb    = alloc((size_t)NN * D_HC);  // reused as SAG 'aggr'
  float*    kb    = alloc((size_t)NN * D_HC);
  float*    vb    = alloc((size_t)NN * D_HC);
  float*    ep    = alloc((size_t)NE * D_HC);
  float*    eaA   = alloc((size_t)NE * D_EF);
  float*    eaB   = alloc((size_t)NE * D_EF);
  float*    alpha = alloc((size_t)NE * 2);
  unsigned* nmax  = (unsigned*)alloc((size_t)NN * 2);
  float*    nsum  = alloc((size_t)NN * 2);
  float*    gstat = alloc(3 * NG);  // cnt | sum | sumsq
  float*    gcnt = gstat, *gsum = gstat + NG, *gss = gstat + 2 * NG;
  float*    sc    = alloc(NN);
  unsigned* gmax  = (unsigned*)alloc(NG);
  float*    gssum = alloc(NG);
  float*    emb   = alloc((size_t)NG * D_HC);
  float*    eagg  = alloc((size_t)NG * D_EF);
  float*    embe  = alloc((size_t)NG * D_HC);

  dim3 B(256);
  int gN128 = cdiv(NN, 128), gE128 = cdiv(NE, 128), gG128 = cdiv(NG, 128);
  int gEw = cdiv(NE, 8), gNw = cdiv(NN, 8);  // wave-per-item grids

  auto gln = [&](const float* X, float* Y, const float* gamma, const float* beta) {
    fill_f32<<<cdiv(3 * NG, 256), B, 0, stream>>>(gstat, 0.f, 3 * NG);
    gln_stats<<<gNw, B, 0, stream>>>(X, batch, gcnt, gsum, gss, NN);
    gln_apply<<<cdiv(NN * D_HC, 256), B, 0, stream>>>(X, Y, batch, gcnt, gsum, gss,
                                                      gamma, beta, 1, NN * D_HC);
  };

  auto run_conv = [&](int i, int j, const float* ea_in) {
    int cj = i * 2 + j;
    const float* wq = qkv_w + (size_t)(cj * 3 + 0) * D_HC * D_HC;
    const float* wk = qkv_w + (size_t)(cj * 3 + 1) * D_HC * D_HC;
    const float* wv = qkv_w + (size_t)(cj * 3 + 2) * D_HC * D_HC;
    const float* bq = qkv_b + (size_t)(cj * 3 + 0) * D_HC;
    const float* bk = qkv_b + (size_t)(cj * 3 + 1) * D_HC;
    const float* bv = qkv_b + (size_t)(cj * 3 + 2) * D_HC;
    gemm_wmma<128, 128, 0><<<gN128, B, 0, stream>>>(h, wq, bq, qb, NN);
    gemm_wmma<128, 128, 0><<<gN128, B, 0, stream>>>(h, wk, bk, kb, NN);
    gemm_wmma<128, 128, 0><<<gN128, B, 0, stream>>>(h, wv, bv, vb, NN);
    // skip connection initializes the output accumulator
    gemm_wmma<128, 128, 0><<<gN128, B, 0, stream>>>(
        h, skip_w + (size_t)cj * D_HC * D_HC, skip_b + (size_t)cj * D_HC, acc, NN);
    gemm_wmma<64, 128, 0><<<gE128, B, 0, stream>>>(
        ea_in, edge_w + (size_t)cj * D_EF * D_HC, nullptr, ep, NE);
    fill_u32<<<cdiv(NN * 2, 256), B, 0, stream>>>(nmax, 0u, NN * 2);
    fill_f32<<<cdiv(NN * 2, 256), B, 0, stream>>>(nsum, 0.f, NN * 2);
    edge_logits<<<gEw, B, 0, stream>>>(qb, kb, ep, src, dst, alpha, nmax, NE);
    edge_exp<<<cdiv(NE * 2, 256), B, 0, stream>>>(alpha, dst, nmax, nsum, NE * 2);
    edge_aggr<<<gEw, B, 0, stream>>>(vb, ep, alpha, nsum, src, dst, acc, NE);
  };

  // ---- stage 0: node encoder + edge encoder ----
  gemm_wmma<64, 128, 0><<<gN128, B, 0, stream>>>(x, w0, b0, acc, NN);
  gln(acc, h, ln0_g, ln0_b);
  gemm_wmma<16, 64, 1><<<gE128, B, 0, stream>>>(edge_attr, we0, be0, eaA, NE);
  float* ea_cur = eaA;
  float* ea_alt = eaB;

  // ---- blocks ----
  for (int i = 0; i < 2; ++i) {
    // conv1 + graph-LN + edge up-projection (ELU)
    run_conv(i, 0, ea_cur);
    gln(acc, h, ng + (size_t)(i * 2 + 0) * D_HC, nbb + (size_t)(i * 2 + 0) * D_HC);
    gemm_wmma<64, 64, 1><<<gE128, B, 0, stream>>>(
        ea_cur, up_w + (size_t)(i * 2 + 0) * D_EF * D_EF,
        up_b + (size_t)(i * 2 + 0) * D_EF, ea_alt, NE);
    { float* t = ea_cur; ea_cur = ea_alt; ea_alt = t; }

    // conv2 (result stays in acc: this is 'h' for SAG pooling) + ea2 (no act)
    run_conv(i, 1, ea_cur);
    gemm_wmma<64, 64, 0><<<gE128, B, 0, stream>>>(
        ea_cur, up_w + (size_t)(i * 2 + 1) * D_EF * D_EF,
        up_b + (size_t)(i * 2 + 1) * D_EF, ea_alt, NE);
    float* ea2 = ea_alt;

    // SAGPooling score = per-graph softmax of GraphConv(aggr, root)
    fill_f32<<<cdiv(NN * D_HC, 256), B, 0, stream>>>(qb, 0.f, NN * D_HC);  // aggr
    node_aggr<<<gEw, B, 0, stream>>>(acc, src, dst, qb, NE);
    fill_u32<<<cdiv(NG, 256), B, 0, stream>>>(gmax, 0u, NG);
    fill_f32<<<cdiv(NG, 256), B, 0, stream>>>(gssum, 0.f, NG);
    node_score<<<gNw, B, 0, stream>>>(qb, acc, sag_wrel + (size_t)i * D_HC, sag_brel + i,
                                      sag_wroot + (size_t)i * D_HC, batch, sc, gmax, NN);
    score_exp<<<cdiv(NN, 256), B, 0, stream>>>(sc, batch, gmax, gssum, NN);
    fill_f32<<<cdiv(NG * D_HC, 256), B, 0, stream>>>(emb, 0.f, NG * D_HC);
    emb_accum<<<cdiv(NN * D_HC, 256), B, 0, stream>>>(acc, sc, gssum, batch, emb, NN * D_HC);

    // edge embedding: segment-sum(ea2 by batch[src]) @ re_w + re_b, ELU
    fill_f32<<<cdiv(NG * D_EF, 256), B, 0, stream>>>(eagg, 0.f, NG * D_EF);
    edge_seg_sum<<<cdiv(NE * D_EF, 256), B, 0, stream>>>(ea2, src, batch, eagg, NE * D_EF);
    gemm_wmma<64, 128, 1><<<gG128, B, 0, stream>>>(
        eagg, re_w + (size_t)i * D_EF * D_HC, re_b + (size_t)i * D_HC, embe, NG);
    emb_finalize<<<NG, dim3(128), 0, stream>>>(emb, embe, out, i == 0 ? 1 : 0);

    // post-block: h = elu(graph_ln(h)), ea = elu(ea2)
    gln(acc, h, ng + (size_t)(i * 2 + 1) * D_HC, nbb + (size_t)(i * 2 + 1) * D_HC);
    elu_inplace<<<cdiv(NE * D_EF, 256), B, 0, stream>>>(ea2, NE * D_EF);
    ea_cur = ea2;
    ea_alt = (ea2 == eaA) ? eaB : eaA;
  }

  (void)in_sizes; (void)n_in; (void)out_size; (void)ws_size;
}
